// GaussKernelAttn_53085795779184
// MI455X (gfx1250) — matrix operand order repaired and checked by reading, not compiled, not run on hardware
//
#include <hip/hip_runtime.h>
#include <math.h>

#define DEV __device__ __forceinline__

typedef __attribute__((ext_vector_type(16))) __bf16          bf16x16;
typedef __attribute__((ext_vector_type(8)))  float           f32x8;
typedef __attribute__((ext_vector_type(8)))  unsigned short  us8;
typedef __attribute__((ext_vector_type(16))) unsigned short  us16;
typedef __attribute__((ext_vector_type(4)))  unsigned int    u32x4;
typedef __attribute__((ext_vector_type(8)))  int             i32x8;
typedef __attribute__((ext_vector_type(4)))  int             i32x4;

// ---- problem constants ----
#define BATCH  8
#define NTOK   1032          // 8 prefix + 32*32 patches
#define NPAD   1056          // 33 j-tiles of 32
#define CDIM   1024
#define NHEAD  16
#define HDIM   64
#define O3     3072
#define MROWS  (BATCH * NTOK)   // 8256 = 516 * 16 (exact)
#define MT     (MROWS / 16)     // 516
#define SCALE  0.125f           // 64^-0.5
#define C2     0.01020408163265306f  // 1/(2*7^2)

// ---------- helpers ----------
DEV unsigned short f2bf(float f) {           // f32 -> bf16, round-to-nearest-even
  unsigned int u = __float_as_uint(f);
  u += 0x7FFFu + ((u >> 16) & 1u);
  return (unsigned short)(u >> 16);
}

DEV f32x8 wmma_bf16(bf16x16 a, bf16x16 b, f32x8 c) {
  return __builtin_amdgcn_wmma_f32_16x16x32_bf16(
      /*neg_a=*/false, a, /*neg_b=*/false, b,
      /*c_mod=*/(short)0, c, /*reuse_a=*/false, /*reuse_b=*/false);
}

DEV bf16x16 pack16(us8 lo, us8 hi) {
  us16 u = __builtin_shufflevector(lo, hi, 0,1,2,3,4,5,6,7,8,9,10,11,12,13,14,15);
  return __builtin_bit_cast(bf16x16, u);
}

// Escape hatch: TDM writes LDS via an integer offset the compiler cannot alias
// back to `tiles`; without this, the LDS fragment loads fold to undef.
DEV void lds_escape(const void* p) {
  asm volatile("" : : "v"(p) : "memory");
}

// A fragment (16x32, MxK) from row-major src[*, ld]; rows row0..row0+15, K chunk k0.
// Lane l: row = l&15; elems 0..7 at K=k0+(l>>4)*8, elems 8..15 at K=k0+16+(l>>4)*8.
DEV bf16x16 load_a(const unsigned short* __restrict__ src, int row0, int rowlim,
                   int ld, int k0, int lane) {
  us8 lo = {}, hi = {};
  const int m = lane & 15, half = lane >> 4;
  const int r = row0 + m;
  if (r < rowlim) {
    const unsigned short* p = src + r * ld + k0 + half * 8;
    lo = *(const us8*)p;
    hi = *(const us8*)(p + 16);
  }
  return pack16(lo, hi);
}

// B fragment (32x16, KxN) from "B-transposed" storage bt[n][k] row-major [*, ld].
// Lane l: col n = l&15; elems 0..7 at K=k0+(l>>4)*8, elems 8..15 at K=k0+16+(l>>4)*8.
DEV bf16x16 load_b(const unsigned short* __restrict__ bt, int n0, int nlim,
                   int ld, int k0, int lane) {
  us8 lo = {}, hi = {};
  const int n = lane & 15, half = lane >> 4;
  const int r = n0 + n;
  if (r < nlim) {
    const unsigned short* p = bt + r * ld + k0 + half * 8;
    lo = *(const us8*)p;
    hi = *(const us8*)(p + 16);
  }
  return pack16(lo, hi);
}

// Tensor Data Mover: 2D tile (tile_rows x tile_k bf16 elems) from global -> LDS.
// D# packing per CDNA5 ISA 8.3/8.4: group0 = {count|flags, lds_addr, gaddr lo,
// gaddr[56:32]|type=2}; group1 = {data_size, tensor_dim0/1 (maxed: tiles divide
// exactly so OOB never triggers), tile_dim0/1, dim0 stride}. groups 2/3 zero (2D).
// This toolchain exposes the 6-arg builtin (extra int32x8 group, then cpol).
DEV void tdm_load_2d(unsigned lds_byte_off, const void* gptr,
                     unsigned tile_k, unsigned tile_rows, unsigned stride_elems) {
  const unsigned long long ga = (unsigned long long)(size_t)gptr;
  u32x4 g0 = {};
  g0[0] = 1u;                                                 // count=1, user D#
  g0[1] = lds_byte_off;                                       // LDS dest (bytes)
  g0[2] = (unsigned)ga;                                       // global_addr[31:0]
  g0[3] = (unsigned)((ga >> 32) & 0x1FFFFFFu) | (2u << 30);   // addr[56:32]|type=2
  i32x8 g1 = {};
  g1[0] = 1 << 16;                                            // data_size = 2 bytes
  g1[1] = (int)0xFFFF0000u;                                   // tensor_dim0 lo16<<16
  g1[2] = (int)0xFFFFFFFFu;                                   // dim0 hi16 | dim1 lo16
  g1[3] = (int)(0x0000FFFFu | (tile_k << 16));                // dim1 hi16 | tile_dim0
  g1[4] = (int)tile_rows;                                     // tile_dim1 (tile_dim2=0)
  g1[5] = (int)stride_elems;                                  // dim0_stride[31:0]
  g1[6] = 0;                                                  // stride0 hi | stride1 lo
  g1[7] = 0;
  i32x4 z4 = {};
  i32x8 z8 = {};
  __builtin_amdgcn_tensor_load_to_lds(g0, g1, z4, z4, z8, 0);
}

// Shared TDM-staged block GEMM: 8 waves compute a 64x32 tile over K=1024.
// Double-buffered LDS: [A buf0 | A buf1 | B buf0 | B buf1] = 24576 bytes.
// gA points at row strip (rowB*CDIM), gB at col strip (colB*CDIM).
DEV f32x8 gemm_block_64x32(const unsigned short* __restrict__ gA,
                           const unsigned short* __restrict__ gB,
                           unsigned short* tiles, int wid, int lane,
                           int msub, int nsub) {
  if (wid == 0) {
    tdm_load_2d(0,     gA, 64, 64, CDIM);
    tdm_load_2d(16384, gB, 64, 32, CDIM);
  }
  f32x8 acc = {};
  for (int kk = 0; kk < 16; ++kk) {           // K = 16 * 64
    const int cur = kk & 1;
    if (wid == 0) {
      if (kk + 1 < 16) {
        const int nxt = cur ^ 1;
        const int kn = (kk + 1) * 64;
        tdm_load_2d((unsigned)(nxt * 8192),         gA + kn, 64, 64, CDIM);
        tdm_load_2d((unsigned)(16384 + nxt * 4096), gB + kn, 64, 32, CDIM);
        __builtin_amdgcn_s_wait_tensorcnt(2);  // in-order: current buffer done
      } else {
        __builtin_amdgcn_s_wait_tensorcnt(0);  // last buffer done
      }
    }
    __syncthreads();
    lds_escape(tiles);   // TDM wrote LDS: keep the ds_loads below alive
    const unsigned short* As = tiles + cur * 4096;          // 64x64 row-major
    const unsigned short* Bs = tiles + 8192 + cur * 2048;   // 32x64 row-major
    acc = wmma_bf16(load_a(As, msub * 16, 1 << 30, 64, 0,  lane),
                    load_b(Bs, nsub * 16, 1 << 30, 64, 0,  lane), acc);
    acc = wmma_bf16(load_a(As, msub * 16, 1 << 30, 64, 32, lane),
                    load_b(Bs, nsub * 16, 1 << 30, 64, 32, lane), acc);
    __syncthreads();   // reads done before TDM overwrites this buffer
  }
  return acc;
}

// analytic gaussian attention addition (zero for prefix rows/cols)
DEV float gauss_add(int i, int j) {
  if (i < 8 || j < 8) return 0.f;
  const int pi = i - 8, pj = j - 8;
  const float di = (float)((pi >> 5) - (pj >> 5));
  const float dj = (float)((pi & 31) - (pj & 31));
  return __expf(-C2 * (di * di + dj * dj));
}

// ---------- kernels ----------
__global__ void cvt_bf16_kernel(const float* __restrict__ s,
                                unsigned short* __restrict__ d, int n) {
  const int i = (blockIdx.x * blockDim.x + threadIdx.x) * 4;
  if (i + 4 <= n) {
    const float4 f = *(const float4*)(s + i);
    d[i + 0] = f2bf(f.x); d[i + 1] = f2bf(f.y);
    d[i + 2] = f2bf(f.z); d[i + 3] = f2bf(f.w);
  }
}

// qkv = x @ qkv_w^T + qkv_b, TDM-staged through LDS.
// Scatter into Q[b,h][n][d], K[b,h][n][d], V^T[b,h][d][n].
__global__ void __launch_bounds__(256)
qkv_gemm_kernel(const unsigned short* __restrict__ xb,
                const unsigned short* __restrict__ wb,
                const float* __restrict__ bias,
                unsigned short* __restrict__ qb,
                unsigned short* __restrict__ kb,
                unsigned short* __restrict__ vt) {
  __shared__ __align__(16) unsigned short tiles[12288];
  const int lane = threadIdx.x & 31;
  const int wid  = threadIdx.x >> 5;          // 0..7
  const int msub = wid & 3, nsub = wid >> 2;  // 4 x 2 sub-tiles
  const int mt4 = blockIdx.x % 129;           // 516/4 block rows
  const int nt2 = blockIdx.x / 129;           // 192/2 block cols
  const int rowB = mt4 * 64, colB = nt2 * 32;
  const int row0 = rowB + msub * 16, col0 = colB + nsub * 16;

  const f32x8 acc = gemm_block_64x32(xb + (size_t)rowB * CDIM,
                                     wb + (size_t)colB * CDIM,
                                     tiles, wid, lane, msub, nsub);

  const int c = lane & 15, hlf = lane >> 4;
  const int col = col0 + c;
  const float bv = bias[col];
#pragma unroll
  for (int r = 0; r < 8; ++r) {
    const int row = row0 + r + 8 * hlf;
    const int bb = row / NTOK;
    const int n  = row - bb * NTOK;
    const unsigned short hv = f2bf(acc[r] + bv);
    if (col < CDIM) {
      const int hh = col >> 6, d = col & 63;
      qb[((bb * NHEAD + hh) * NTOK + n) * HDIM + d] = hv;
    } else if (col < 2 * CDIM) {
      const int cc = col - CDIM, hh = cc >> 6, d = cc & 63;
      kb[((bb * NHEAD + hh) * NTOK + n) * HDIM + d] = hv;
    } else {
      const int cc = col - 2 * CDIM, hh = cc >> 6, d = cc & 63;
      vt[((bb * NHEAD + hh) * HDIM + d) * NPAD + n] = hv;
    }
  }
}

// flash attention with gaussian score addition; one wave per (b, h, 16-row i-tile)
__global__ void __launch_bounds__(32)
attn_kernel(const unsigned short* __restrict__ qb,
            const unsigned short* __restrict__ kb,
            const unsigned short* __restrict__ vt,
            unsigned short* __restrict__ ctx) {
  __shared__ __align__(16) unsigned short plds[16 * 40];  // P tile bounce buffer
  const int lane = threadIdx.x;
  const int it = blockIdx.x;   // 0..64
  const int h  = blockIdx.y;
  const int b  = blockIdx.z;
  const unsigned short* Q  = qb + (b * NHEAD + h) * NTOK * HDIM;
  const unsigned short* K  = kb + (b * NHEAD + h) * NTOK * HDIM;
  const unsigned short* VT = vt + (b * NHEAD + h) * HDIM * NPAD;
  const int i0 = it * 16;
  const int c = lane & 15, hlf = lane >> 4;

  const bf16x16 qa0 = load_a(Q, i0, NTOK, HDIM, 0,  lane);   // reused every j step
  const bf16x16 qa1 = load_a(Q, i0, NTOK, HDIM, 32, lane);

  f32x8 o0 = {}, o1 = {}, o2 = {}, o3 = {};
  float mrow[8], lrow[8];
#pragma unroll
  for (int r = 0; r < 8; ++r) { mrow[r] = -1e30f; lrow[r] = 0.f; }

  for (int j0 = 0; j0 < NPAD; j0 += 32) {
    // S tile: 16 x 32 (two 16x16 C fragments), K-dim = HDIM in 2 chunks of 32
    f32x8 s0 = {}, s1 = {};
    s0 = wmma_bf16(qa0, load_b(K, j0,      NTOK, HDIM, 0,  lane), s0);
    s0 = wmma_bf16(qa1, load_b(K, j0,      NTOK, HDIM, 32, lane), s0);
    s1 = wmma_bf16(qa0, load_b(K, j0 + 16, NTOK, HDIM, 0,  lane), s1);
    s1 = wmma_bf16(qa1, load_b(K, j0 + 16, NTOK, HDIM, 32, lane), s1);

    const int ja = j0 + c, jb = j0 + 16 + c;
#pragma unroll
    for (int r = 0; r < 8; ++r) {
      const int i = i0 + r + 8 * hlf;
      float sa = s0[r] * SCALE + gauss_add(i, ja);
      float sb = s1[r] * SCALE + gauss_add(i, jb);
      if (ja >= NTOK) sa = -INFINITY;
      if (jb >= NTOK) sb = -INFINITY;
      // row max across the 16 lanes holding this row's columns
      float mx = fmaxf(sa, sb);
      mx = fmaxf(mx, __shfl_xor(mx, 1, 32));
      mx = fmaxf(mx, __shfl_xor(mx, 2, 32));
      mx = fmaxf(mx, __shfl_xor(mx, 4, 32));
      mx = fmaxf(mx, __shfl_xor(mx, 8, 32));
      const float mn = fmaxf(mrow[r], mx);
      const float alpha = __expf(mrow[r] - mn);
      mrow[r] = mn;
      const float pa = __expf(sa - mn);
      const float pb = __expf(sb - mn);
      float rs = pa + pb;
      rs += __shfl_xor(rs, 1, 32);
      rs += __shfl_xor(rs, 2, 32);
      rs += __shfl_xor(rs, 4, 32);
      rs += __shfl_xor(rs, 8, 32);
      lrow[r] = lrow[r] * alpha + rs;
      o0[r] *= alpha; o1[r] *= alpha; o2[r] *= alpha; o3[r] *= alpha;
      const int prow = r + 8 * hlf;
      plds[prow * 40 + c]      = f2bf(pa);
      plds[prow * 40 + 16 + c] = f2bf(pb);
    }
    __syncthreads();
    // reload P in A-fragment layout (contiguous b128 LDS reads)
    const unsigned short* pp = plds + (lane & 15) * 40 + hlf * 8;
    const bf16x16 pf = pack16(*(const us8*)pp, *(const us8*)(pp + 16));
    // O += P x V  (V^T rows are contiguous K runs; pad cols multiply zero P)
    o0 = wmma_bf16(pf, load_b(VT, 0,  64, NPAD, j0, lane), o0);
    o1 = wmma_bf16(pf, load_b(VT, 16, 64, NPAD, j0, lane), o1);
    o2 = wmma_bf16(pf, load_b(VT, 32, 64, NPAD, j0, lane), o2);
    o3 = wmma_bf16(pf, load_b(VT, 48, 64, NPAD, j0, lane), o3);
    __syncthreads();
  }

#pragma unroll
  for (int r = 0; r < 8; ++r) {
    const int i = i0 + r + 8 * hlf;
    if (i < NTOK) {
      const float inv = 1.0f / lrow[r];
      const int base = (b * NTOK + i) * CDIM + h * HDIM;
      ctx[base +  0 + c] = f2bf(o0[r] * inv);
      ctx[base + 16 + c] = f2bf(o1[r] * inv);
      ctx[base + 32 + c] = f2bf(o2[r] * inv);
      ctx[base + 48 + c] = f2bf(o3[r] * inv);
    }
  }
}

// out = ctx @ proj_w^T + proj_b (f32 output), same TDM-staged block GEMM
__global__ void __launch_bounds__(256)
proj_gemm_kernel(const unsigned short* __restrict__ cx,
                 const unsigned short* __restrict__ wb,
                 const float* __restrict__ bias,
                 float* __restrict__ out) {
  __shared__ __align__(16) unsigned short tiles[12288];
  const int lane = threadIdx.x & 31;
  const int wid  = threadIdx.x >> 5;
  const int msub = wid & 3, nsub = wid >> 2;
  const int mt4 = blockIdx.x % 129;           // 516/4 block rows
  const int nt2 = blockIdx.x / 129;           // 64/2  block cols
  const int rowB = mt4 * 64, colB = nt2 * 32;
  const int row0 = rowB + msub * 16, col0 = colB + nsub * 16;

  const f32x8 acc = gemm_block_64x32(cx + (size_t)rowB * CDIM,
                                     wb + (size_t)colB * CDIM,
                                     tiles, wid, lane, msub, nsub);

  const int c = lane & 15, hlf = lane >> 4;
  const int col = col0 + c;
  const float bv = bias[col];
#pragma unroll
  for (int r = 0; r < 8; ++r) {
    const int row = row0 + r + 8 * hlf;
    out[row * CDIM + col] = acc[r] + bv;
  }
}

// ---------- launch ----------
extern "C" void kernel_launch(void* const* d_in, const int* in_sizes, int n_in,
                              void* d_out, int out_size, void* d_ws, size_t ws_size,
                              hipStream_t stream) {
  (void)in_sizes; (void)n_in; (void)out_size; (void)ws_size;
  const float* x      = (const float*)d_in[0];
  const float* qkv_w  = (const float*)d_in[1];
  const float* qkv_b  = (const float*)d_in[2];
  const float* proj_w = (const float*)d_in[3];
  const float* proj_b = (const float*)d_in[4];
  float* out = (float*)d_out;

  char* ws = (char*)d_ws;
  size_t off = 0;
  unsigned short* xb = (unsigned short*)(ws + off); off += (size_t)MROWS * CDIM * 2;
  unsigned short* wq = (unsigned short*)(ws + off); off += (size_t)O3 * CDIM * 2;
  unsigned short* wp = (unsigned short*)(ws + off); off += (size_t)CDIM * CDIM * 2;
  unsigned short* qb = (unsigned short*)(ws + off); off += (size_t)BATCH * NHEAD * NTOK * HDIM * 2;
  unsigned short* kb = (unsigned short*)(ws + off); off += (size_t)BATCH * NHEAD * NTOK * HDIM * 2;
  unsigned short* vt = (unsigned short*)(ws + off); off += (size_t)BATCH * NHEAD * HDIM * NPAD * 2;
  unsigned short* cx = (unsigned short*)(ws + off); off += (size_t)MROWS * CDIM * 2;

  int n = MROWS * CDIM;
  cvt_bf16_kernel<<<dim3((n / 4 + 255) / 256), dim3(256), 0, stream>>>(x, xb, n);
  n = O3 * CDIM;
  cvt_bf16_kernel<<<dim3((n / 4 + 255) / 256), dim3(256), 0, stream>>>(qkv_w, wq, n);
  n = CDIM * CDIM;
  cvt_bf16_kernel<<<dim3((n / 4 + 255) / 256), dim3(256), 0, stream>>>(proj_w, wp, n);

  // blocks = block-rows x block-cols, each block = 64x32 output tile
  qkv_gemm_kernel<<<dim3(129 * 96), dim3(256), 0, stream>>>(xb, wq, qkv_b, qb, kb, vt);
  attn_kernel<<<dim3(65, NHEAD, BATCH), dim3(32), 0, stream>>>(qb, kb, vt, cx);
  proj_gemm_kernel<<<dim3(129 * 32), dim3(256), 0, stream>>>(cx, wp, proj_b, out);
}
